// ST_GNN_52347061403821
// MI455X (gfx1250) — matrix operand order repaired and checked by reading, not compiled, not run on hardware
//
#include <hip/hip_runtime.h>
#include <hip/hip_bf16.h>
#include <math.h>
#include <cstddef>
#include <cstdint>

// ---------------------------------------------------------------------------
// Types for CDNA5 WMMA (gfx1250, wave32)
// ---------------------------------------------------------------------------
typedef __bf16 bf16_t;
typedef bf16_t bf16x8_t  __attribute__((ext_vector_type(8)));
typedef bf16_t bf16x16_t __attribute__((ext_vector_type(16)));
typedef float  f32x8_t   __attribute__((ext_vector_type(8)));

#define NN_NODES 20000
#define EE_EDGES 320000
#define HH 4
#define CC 128
#define DD 512

// ---------------------------------------------------------------------------
// Helpers
// ---------------------------------------------------------------------------
__device__ __forceinline__ void atomicMaxF(float* addr, float v) {
    if (v >= 0.0f) atomicMax((int*)addr, __float_as_int(v));
    else           atomicMin((unsigned int*)addr, __float_as_uint(v));
}
__device__ __forceinline__ float sigmoidf_(float x) { return 1.0f / (1.0f + __expf(-x)); }

// ---------------------------------------------------------------------------
// Weight transpose + f32->bf16 convert:  W[K,N] row-major -> Wt[N,K] bf16
// ---------------------------------------------------------------------------
__global__ void k_transpose_bf16(const float* __restrict__ W, bf16_t* __restrict__ Wt,
                                 int K, int N) {
    int idx = blockIdx.x * 256 + threadIdx.x;
    if (idx >= K * N) return;
    int n = idx / K, k = idx % K;
    Wt[idx] = (bf16_t)W[(size_t)k * N + n];
}

__global__ void k_cvt_bf16(const float* __restrict__ in, bf16_t* __restrict__ out, size_t count) {
    size_t i = (size_t)blockIdx.x * 256 + threadIdx.x;
    if (i < count) out[i] = (bf16_t)in[i];
}

__global__ void k_fill_f32(float* __restrict__ p, float v, size_t count) {
    size_t i = (size_t)blockIdx.x * 256 + threadIdx.x;
    if (i < count) p[i] = v;
}

// agg[n, 0:512] = gat_bias[l][0:512]
__global__ void k_init_agg(float* __restrict__ agg, const float* __restrict__ bias, int Nn) {
    size_t i = (size_t)blockIdx.x * 256 + threadIdx.x;
    if (i < (size_t)Nn * DD) agg[i] = bias[i & (DD - 1)];
}

// ---------------------------------------------------------------------------
// WMMA GEMM:  C[M,N] = A[M,K] @ Bt[N,K]^T   (A,Bt bf16 row-major; C f32)
// Optional bias (per output column) and ReLU epilogue.
// Block: 256 thr = 8 waves, tile 128(M) x 64(N), K step 32.
// Double-buffered LDS fed by GLOBAL_LOAD_ASYNC_TO_LDS_B128 (ASYNCcnt):
// tile k+1 streams into the other buffer while WMMAs consume tile k;
// a single s_wait_asynccnt + barrier per K-step.
// ---------------------------------------------------------------------------
#define LDA 40   // padded LDS row stride (bf16 elems): conflict-free & 16B aligned

__global__ __launch_bounds__(256)
void k_gemm_bf16(const bf16_t* __restrict__ A, const bf16_t* __restrict__ Bt,
                 float* __restrict__ C, int M, int N, int K,
                 const float* __restrict__ bias, int act) {
    __shared__ __align__(16) bf16_t sA[2][128 * LDA];
    __shared__ __align__(16) bf16_t sB[2][64 * LDA];

    const int tid  = threadIdx.x;
    const int wave = tid >> 5;
    const int lane = tid & 31;
    const int g    = lane >> 4;     // half-wave group
    const int ln   = lane & 15;
    const int m0   = blockIdx.x * 128;
    const int n0   = blockIdx.y * 64;

    // ---- per-thread async-copy assignments -------------------------------
    // A tile: 128 rows x 32 bf16; 2 threads/row, 2 x 16B chunks each.
    const int arow  = tid >> 1;
    const int ahalf = tid & 1;
    int am = m0 + arow;
    if (am >= M) am = M - 1;   // clamp: rows >= M are masked in the epilogue
    const unsigned aGoff = (unsigned)(((size_t)am * K + ahalf * 16) * sizeof(bf16_t));
    // B tile: 64 rows x 32 bf16; 4 threads/row, 1 x 16B chunk each.
    const int brow  = tid >> 2;
    const int bpart = tid & 3;
    const unsigned bGoff = (unsigned)(((size_t)(n0 + brow) * K + bpart * 8) * sizeof(bf16_t));

    const unsigned aLds[2] = {
        (unsigned)(uintptr_t)&sA[0][arow * LDA + ahalf * 16],
        (unsigned)(uintptr_t)&sA[1][arow * LDA + ahalf * 16] };
    const unsigned bLds[2] = {
        (unsigned)(uintptr_t)&sB[0][brow * LDA + bpart * 8],
        (unsigned)(uintptr_t)&sB[1][brow * LDA + bpart * 8] };

    auto issue_tile = [&](int k0, int buf) {
        unsigned kb = (unsigned)(k0 * (int)sizeof(bf16_t));
        unsigned ao = aGoff + kb;
        unsigned bo = bGoff + kb;
        // instruction offset applies to BOTH the global and the LDS address,
        // so the second 16B chunk of an A row reuses the same registers.
        asm volatile("global_load_async_to_lds_b128 %0, %1, %2"
                     :: "v"(aLds[buf]), "v"(ao), "s"(A) : "memory");
        asm volatile("global_load_async_to_lds_b128 %0, %1, %2 offset:16"
                     :: "v"(aLds[buf]), "v"(ao), "s"(A) : "memory");
        asm volatile("global_load_async_to_lds_b128 %0, %1, %2"
                     :: "v"(bLds[buf]), "v"(bo), "s"(Bt) : "memory");
    };

    f32x8_t acc[4] = {};

    // ---- prologue: stream tile 0 into buffer 0
    issue_tile(0, 0);
    asm volatile("s_wait_asynccnt 0x0" ::: "memory");
    __syncthreads();

    int p = 0;
    for (int k0 = 0; k0 < K; k0 += 32) {
        const int kn = k0 + 32;
        if (kn < K) issue_tile(kn, p ^ 1);   // overlap next tile with compute

        // --- A fragment (16x32): elems 0..7 = A[m][g*8+0..7], 8..15 = A[m][16+g*8+..]
        const bf16_t* aRow = &sA[p][(wave * 16 + ln) * LDA + g * 8];
        bf16x8_t a_lo = *(const bf16x8_t*)(aRow);
        bf16x8_t a_hi = *(const bf16x8_t*)(aRow + 16);
        bf16x16_t afrag;
        #pragma unroll
        for (int i = 0; i < 8; ++i) { afrag[i] = a_lo[i]; afrag[8 + i] = a_hi[i]; }

        #pragma unroll
        for (int nt = 0; nt < 4; ++nt) {
            // B fragment (32x16): elems 0..7 = Bt[n][g*8+0..7], 8..15 = Bt[n][16+g*8+..]
            const bf16_t* bRow = &sB[p][(nt * 16 + ln) * LDA + g * 8];
            bf16x8_t b_lo = *(const bf16x8_t*)(bRow);
            bf16x8_t b_hi = *(const bf16x8_t*)(bRow + 16);
            bf16x16_t bfrag;
            #pragma unroll
            for (int i = 0; i < 8; ++i) { bfrag[i] = b_lo[i]; bfrag[8 + i] = b_hi[i]; }

            acc[nt] = __builtin_amdgcn_wmma_f32_16x16x32_bf16(
                false, afrag, false, bfrag, (short)0, acc[nt], false, false);
        }

        // next buffer's async copies must be complete; __syncthreads drains
        // DScnt so this buffer may be overwritten next iteration.
        asm volatile("s_wait_asynccnt 0x0" ::: "memory");
        __syncthreads();
        p ^= 1;
    }

    // --- epilogue: C/D layout -> row = v + g*8, col = ln
    #pragma unroll
    for (int nt = 0; nt < 4; ++nt) {
        #pragma unroll
        for (int v = 0; v < 8; ++v) {
            int m = m0 + wave * 16 + g * 8 + v;
            int n = n0 + nt * 16 + ln;
            if (m < M) {
                float val = acc[nt][v];
                if (bias) val += bias[n];
                if (act == 1) val = fmaxf(val, 0.0f);
                C[(size_t)m * N + n] = val;
            }
        }
    }
}

// ---------------------------------------------------------------------------
// Node encoder: x[n,c] = vf[n,:16] @ W[16,128] + b    (x stored [N,128])
// ---------------------------------------------------------------------------
__global__ __launch_bounds__(256)
void k_node_enc(const float* __restrict__ vf, const float* __restrict__ W,
                const float* __restrict__ b, float* __restrict__ x, int Nn) {
    __shared__ float sW[16 * 128];
    __shared__ float sb[128];
    int tid = threadIdx.x;
    for (int i = tid; i < 16 * 128; i += 256) sW[i] = W[i];
    if (tid < 128) sb[tid] = b[tid];
    __syncthreads();
    int idx = blockIdx.x * 256 + tid;
    if (idx >= Nn * 128) return;
    int n = idx >> 7, c = idx & 127;
    float a = sb[c];
    #pragma unroll
    for (int k = 0; k < 16; ++k) a += vf[(size_t)n * 16 + k] * sW[k * 128 + c];
    x[(size_t)n * 128 + c] = a;
}

// ---------------------------------------------------------------------------
// Edge encoder fused with per-layer attention-edge projection:
//   e = edge_attr @ encW + encb               [128]   (kept in LDS only)
//   e_att[l][e][h] = e . att_edge_W[l][:,h]   for all 4 layers, 4 heads
// ---------------------------------------------------------------------------
__global__ __launch_bounds__(128)
void k_edge_enc_att(const float* __restrict__ edge_attr, const float* __restrict__ encW,
                    const float* __restrict__ encb, const float* __restrict__ aeW,
                    float* __restrict__ e_att, int E_) {
    __shared__ float sW[8 * 128];
    __shared__ float sb[128];
    __shared__ float sAe[4 * 128 * 4];
    __shared__ float sRow[128];
    __shared__ float sPart[16][8];
    int tid = threadIdx.x;
    for (int i = tid; i < 8 * 128; i += 128) sW[i] = encW[i];
    sb[tid] = encb[tid];
    for (int i = tid; i < 4 * 128 * 4; i += 128) sAe[i] = aeW[i];
    __syncthreads();

    int e = blockIdx.x;
    float a = sb[tid];
    #pragma unroll
    for (int k = 0; k < 8; ++k) a += edge_attr[(size_t)e * 8 + k] * sW[k * 128 + tid];
    sRow[tid] = a;
    __syncthreads();

    int combo = tid >> 3, part = tid & 7;        // 16 (l,h) combos x 8 partials
    int l = combo >> 2, h = combo & 3;
    float p = 0.0f;
    for (int c = part * 16; c < part * 16 + 16; ++c)
        p += sRow[c] * sAe[l * 512 + c * 4 + h];
    sPart[combo][part] = p;
    __syncthreads();
    if (tid < 16) {
        float s = 0.0f;
        #pragma unroll
        for (int j = 0; j < 8; ++j) s += sPart[tid][j];
        int l2 = tid >> 2, h2 = tid & 3;
        e_att[((size_t)l2 * E_ + e) * 4 + h2] = s;
    }
}

// ---------------------------------------------------------------------------
// Per-node attention logits: s_n[n,h] = h[n,h,:].att_src[h],  d_n likewise
// ---------------------------------------------------------------------------
__global__ __launch_bounds__(256)
void k_node_logits(const float* __restrict__ hfeat, const float* __restrict__ asrc,
                   const float* __restrict__ adst, float* __restrict__ s_n,
                   float* __restrict__ d_n, int Nn) {
    __shared__ float sS[512];
    __shared__ float sD[512];
    int tid = threadIdx.x;
    for (int i = tid; i < 512; i += 256) { sS[i] = asrc[i]; sD[i] = adst[i]; }
    __syncthreads();
    int idx = blockIdx.x * 256 + tid;
    if (idx >= Nn * 4) return;
    int n = idx >> 2, h = idx & 3;
    const float* hp = hfeat + (size_t)n * DD + h * CC;
    float s = 0.0f, d = 0.0f;
    #pragma unroll 8
    for (int c = 0; c < CC; ++c) {
        float hv = hp[c];
        s += hv * sS[h * CC + c];
        d += hv * sD[h * CC + c];
    }
    s_n[idx] = s;
    d_n[idx] = d;
}

// ---------------------------------------------------------------------------
// Softmax pass A: raw logits + leaky-relu; atomic float-max into m[dst,h]
// ---------------------------------------------------------------------------
__global__ __launch_bounds__(256)
void k_att_a(const float* __restrict__ s_n, const float* __restrict__ d_n,
             const float* __restrict__ eatt_l, const int* __restrict__ src,
             const int* __restrict__ dst, float* __restrict__ alpha,
             float* __restrict__ mbuf, int E_) {
    int idx = blockIdx.x * 256 + threadIdx.x;
    if (idx >= E_ * 4) return;
    int e = idx >> 2, h = idx & 3;
    float a = s_n[src[e] * 4 + h] + d_n[dst[e] * 4 + h] + eatt_l[idx];
    a = (a > 0.0f) ? a : 0.2f * a;            // leaky relu, slope 0.2
    alpha[idx] = a;
    atomicMaxF(&mbuf[dst[e] * 4 + h], a);
}

// Softmax pass B: ex = exp(a - m[dst]); denom[dst] += ex  (in-place over alpha)
__global__ __launch_bounds__(256)
void k_att_b(float* __restrict__ alpha, const float* __restrict__ mbuf,
             float* __restrict__ denom, const int* __restrict__ dst, int E_) {
    int idx = blockIdx.x * 256 + threadIdx.x;
    if (idx >= E_ * 4) return;
    int e = idx >> 2, h = idx & 3;
    float ex = __expf(alpha[idx] - mbuf[dst[e] * 4 + h]);
    alpha[idx] = ex;
    atomicAdd(&denom[dst[e] * 4 + h], ex);
}

// ---------------------------------------------------------------------------
// Weighted message scatter: agg[dst, :] += h[src, :] * (ex/denom)[edge, head]
// One wave (32 lanes) per edge; lane handles 16 strided channels.
// ---------------------------------------------------------------------------
__global__ __launch_bounds__(256)
void k_scatter(const float* __restrict__ hfeat, const float* __restrict__ ex,
               const float* __restrict__ denom, const int* __restrict__ src,
               const int* __restrict__ dst, float* __restrict__ agg, int E_) {
    int e = blockIdx.x * 8 + (threadIdx.x >> 5);
    if (e >= E_) return;
    int lane = threadIdx.x & 31;
    int s = src[e], d = dst[e];
    float w[4];
    #pragma unroll
    for (int h = 0; h < 4; ++h)
        w[h] = ex[(size_t)e * 4 + h] / (denom[(size_t)d * 4 + h] + 1e-16f);
    #pragma unroll
    for (int j = 0; j < 16; ++j) {
        int ch = lane + 32 * j;
        float v = hfeat[(size_t)s * DD + ch] * w[ch >> 7];
        atomicAdd(&agg[(size_t)d * DD + ch], v);
    }
}

// ---------------------------------------------------------------------------
// Fused GRU (h_prev == 0 -> gh = b_hh) + LayerNorm.  One block per node.
// ---------------------------------------------------------------------------
__global__ __launch_bounds__(256)
void k_gru_ln(const float* __restrict__ gx, const float* __restrict__ b_ih,
              const float* __restrict__ b_hh, const float* __restrict__ lng,
              const float* __restrict__ lnb, float* __restrict__ xout) {
    __shared__ float red1[256];
    __shared__ float red2[256];
    __shared__ float s_mu, s_rs;
    int n = blockIdx.x, tid = threadIdx.x;
    const float* gxr = gx + (size_t)n * 1536;

    float hv[2];
    #pragma unroll
    for (int q = 0; q < 2; ++q) {
        int c = tid + q * 256;
        float r  = sigmoidf_(gxr[c]        + b_ih[c]        + b_hh[c]);
        float z  = sigmoidf_(gxr[512 + c]  + b_ih[512 + c]  + b_hh[512 + c]);
        float nn = tanhf(gxr[1024 + c] + b_ih[1024 + c] + r * b_hh[1024 + c]);
        hv[q] = (1.0f - z) * nn;               // + z * 0
    }
    red1[tid] = hv[0] + hv[1];
    red2[tid] = hv[0] * hv[0] + hv[1] * hv[1];
    __syncthreads();
    for (int s = 128; s > 0; s >>= 1) {
        if (tid < s) { red1[tid] += red1[tid + s]; red2[tid] += red2[tid + s]; }
        __syncthreads();
    }
    if (tid == 0) {
        float mu  = red1[0] * (1.0f / 512.0f);
        float var = red2[0] * (1.0f / 512.0f) - mu * mu;
        s_mu = mu;
        s_rs = rsqrtf(var + 1e-5f);
    }
    __syncthreads();
    #pragma unroll
    for (int q = 0; q < 2; ++q) {
        int c = tid + q * 256;
        xout[(size_t)n * DD + c] = (hv[q] - s_mu) * s_rs * lng[c] + lnb[c];
    }
}

// ---------------------------------------------------------------------------
// Risk head final: out[n,512] = sigmoid(dot(t1[n,:128], W2) + b2)
// ---------------------------------------------------------------------------
__global__ __launch_bounds__(128)
void k_risk_head(const float* __restrict__ t1, const float* __restrict__ W2,
                 const float* __restrict__ b2, float* __restrict__ out) {
    __shared__ float red[128];
    int n = blockIdx.x, tid = threadIdx.x;
    red[tid] = t1[(size_t)n * 128 + tid] * W2[tid];
    __syncthreads();
    for (int s = 64; s > 0; s >>= 1) {
        if (tid < s) red[tid] += red[tid + s];
        __syncthreads();
    }
    if (tid == 0) out[(size_t)n * 577 + 512] = sigmoidf_(red[0] + b2[0]);
}

// out[n, 0:512] = x ; out[n, 513:577] = coll  (slot 512 written by risk head)
__global__ void k_assemble(const float* __restrict__ x, const float* __restrict__ coll,
                           float* __restrict__ out, int Nn) {
    size_t idx = (size_t)blockIdx.x * 256 + threadIdx.x;
    if (idx >= (size_t)Nn * 577) return;
    int n = (int)(idx / 577), c = (int)(idx % 577);
    if (c < 512)      out[idx] = x[(size_t)n * DD + c];
    else if (c > 512) out[idx] = coll[(size_t)n * 64 + (c - 513)];
}

// ---------------------------------------------------------------------------
// Host orchestration
// ---------------------------------------------------------------------------
extern "C" void kernel_launch(void* const* d_in, const int* in_sizes, int n_in,
                              void* d_out, int out_size, void* d_ws, size_t ws_size,
                              hipStream_t stream) {
    const int NN = NN_NODES, EE = EE_EDGES, L = 4;

    const float* vf        = (const float*)d_in[0];
    const int*   eidx      = (const int*)  d_in[1];
    const float* eattr     = (const float*)d_in[2];
    const float* enc_nW    = (const float*)d_in[3];
    const float* enc_nb    = (const float*)d_in[4];
    const float* enc_eW    = (const float*)d_in[5];
    const float* enc_eb    = (const float*)d_in[6];
    const float* lin_W0    = (const float*)d_in[7];
    const float* lin_Wr    = (const float*)d_in[8];
    const float* att_src   = (const float*)d_in[9];
    const float* att_dst   = (const float*)d_in[10];
    const float* att_eW    = (const float*)d_in[11];
    const float* gat_bias  = (const float*)d_in[12];
    const float* gru_Wih   = (const float*)d_in[13];
    /* gru_W_hh (d_in[14]) unused: h_prev == 0 */
    const float* gru_bih   = (const float*)d_in[15];
    const float* gru_bhh   = (const float*)d_in[16];
    const float* ln_g      = (const float*)d_in[17];
    const float* ln_b      = (const float*)d_in[18];
    const float* risk_W1   = (const float*)d_in[19];
    const float* risk_b1   = (const float*)d_in[20];
    const float* risk_W2   = (const float*)d_in[21];
    const float* risk_b2   = (const float*)d_in[22];
    const float* coll_W1   = (const float*)d_in[23];
    const float* coll_b1   = (const float*)d_in[24];
    const float* coll_W2   = (const float*)d_in[25];
    const float* coll_b2   = (const float*)d_in[26];
    float* out = (float*)d_out;

    const int* src = eidx;
    const int* dst = eidx + EE;

    // ---- workspace layout (deterministic every call)
    char* base = (char*)d_ws;
    size_t off = 0;
    auto alloc = [&](size_t bytes) -> char* {
        char* p = base + off;
        off += (bytes + 255) & ~(size_t)255;
        return p;
    };
    bf16_t* wt_lin0 = (bf16_t*)alloc((size_t)512 * 128 * 2);
    bf16_t* wt_lin  = (bf16_t*)alloc((size_t)3 * 512 * 512 * 2);
    bf16_t* wt_gru  = (bf16_t*)alloc((size_t)4 * 1536 * 512 * 2);
    bf16_t* wt_r1   = (bf16_t*)alloc((size_t)128 * 512 * 2);
    bf16_t* wt_c1   = (bf16_t*)alloc((size_t)128 * 512 * 2);
    bf16_t* wt_c2   = (bf16_t*)alloc((size_t)64 * 128 * 2);
    float*  x_f32   = (float*) alloc((size_t)NN * DD * 4);
    bf16_t* x_bf    = (bf16_t*)alloc((size_t)NN * DD * 2);
    float*  hfeat   = (float*) alloc((size_t)NN * DD * 4);
    float*  eatt    = (float*) alloc((size_t)L * EE * 4 * 4);
    float*  s_n     = (float*) alloc((size_t)NN * 4 * 4);
    float*  d_n     = (float*) alloc((size_t)NN * 4 * 4);
    float*  mbuf    = (float*) alloc((size_t)NN * 4 * 4);
    float*  denom   = (float*) alloc((size_t)NN * 4 * 4);
    float*  exb     = (float*) alloc((size_t)EE * 4 * 4);
    float*  agg     = (float*) alloc((size_t)NN * DD * 4);
    bf16_t* g_bf    = (bf16_t*)alloc((size_t)NN * DD * 2);
    float*  gx      = (float*) alloc((size_t)NN * 1536 * 4);
    float*  t1      = (float*) alloc((size_t)NN * 128 * 4);
    float*  t2      = (float*) alloc((size_t)NN * 128 * 4);
    bf16_t* t2bf    = (bf16_t*)alloc((size_t)NN * 128 * 2);
    float*  collb   = (float*) alloc((size_t)NN * 64 * 4);
    (void)ws_size; (void)n_in; (void)in_sizes; (void)out_size;

    auto cdiv = [](size_t a, size_t b) { return (int)((a + b - 1) / b); };
    const int MB = cdiv(NN, 128);   // 157 row-tiles

    // ---- weight pre-transposition to bf16 [Nout, K]
    k_transpose_bf16<<<cdiv(128 * 512, 256), 256, 0, stream>>>(lin_W0, wt_lin0, 128, 512);
    for (int i = 0; i < 3; ++i)
        k_transpose_bf16<<<cdiv(512 * 512, 256), 256, 0, stream>>>(
            lin_Wr + (size_t)i * 512 * 512, wt_lin + (size_t)i * 512 * 512, 512, 512);
    for (int l = 0; l < L; ++l)
        k_transpose_bf16<<<cdiv(512 * 1536, 256), 256, 0, stream>>>(
            gru_Wih + (size_t)l * 512 * 1536, wt_gru + (size_t)l * 1536 * 512, 512, 1536);
    k_transpose_bf16<<<cdiv(512 * 128, 256), 256, 0, stream>>>(risk_W1, wt_r1, 512, 128);
    k_transpose_bf16<<<cdiv(512 * 128, 256), 256, 0, stream>>>(coll_W1, wt_c1, 512, 128);
    k_transpose_bf16<<<cdiv(128 * 64, 256), 256, 0, stream>>>(coll_W2, wt_c2, 128, 64);

    // ---- encoders
    k_node_enc<<<cdiv((size_t)NN * 128, 256), 256, 0, stream>>>(vf, enc_nW, enc_nb, x_f32, NN);
    k_edge_enc_att<<<EE, 128, 0, stream>>>(eattr, enc_eW, enc_eb, att_eW, eatt, EE);

    // ---- layers
    int curK = 128;
    for (int l = 0; l < L; ++l) {
        k_cvt_bf16<<<cdiv((size_t)NN * curK, 256), 256, 0, stream>>>(x_f32, x_bf, (size_t)NN * curK);

        const bf16_t* W = (l == 0) ? wt_lin0 : wt_lin + (size_t)(l - 1) * 512 * 512;
        k_gemm_bf16<<<dim3(MB, 8), 256, 0, stream>>>(x_bf, W, hfeat, NN, 512, curK, nullptr, 0);

        k_node_logits<<<cdiv((size_t)NN * 4, 256), 256, 0, stream>>>(
            hfeat, att_src + (size_t)l * 512, att_dst + (size_t)l * 512, s_n, d_n, NN);

        k_fill_f32<<<cdiv((size_t)NN * 4, 256), 256, 0, stream>>>(mbuf, -INFINITY, (size_t)NN * 4);
        k_fill_f32<<<cdiv((size_t)NN * 4, 256), 256, 0, stream>>>(denom, 0.0f, (size_t)NN * 4);

        k_att_a<<<cdiv((size_t)EE * 4, 256), 256, 0, stream>>>(
            s_n, d_n, eatt + (size_t)l * EE * 4, src, dst, exb, mbuf, EE);
        k_att_b<<<cdiv((size_t)EE * 4, 256), 256, 0, stream>>>(exb, mbuf, denom, dst, EE);

        k_init_agg<<<cdiv((size_t)NN * DD, 256), 256, 0, stream>>>(agg, gat_bias + (size_t)l * DD, NN);
        k_scatter<<<cdiv(EE, 8), 256, 0, stream>>>(hfeat, exb, denom, src, dst, agg, EE);

        k_cvt_bf16<<<cdiv((size_t)NN * DD, 256), 256, 0, stream>>>(agg, g_bf, (size_t)NN * DD);
        k_gemm_bf16<<<dim3(MB, 24), 256, 0, stream>>>(
            g_bf, wt_gru + (size_t)l * 1536 * 512, gx, NN, 1536, 512, nullptr, 0);

        k_gru_ln<<<NN, 256, 0, stream>>>(gx, gru_bih + (size_t)l * 1536,
                                         gru_bhh + (size_t)l * 1536,
                                         ln_g + (size_t)l * DD, ln_b + (size_t)l * DD, x_f32);
        curK = 512;
    }

    // ---- heads
    k_cvt_bf16<<<cdiv((size_t)NN * DD, 256), 256, 0, stream>>>(x_f32, x_bf, (size_t)NN * DD);
    k_gemm_bf16<<<dim3(MB, 2), 256, 0, stream>>>(x_bf, wt_r1, t1, NN, 128, 512, risk_b1, 1);
    k_gemm_bf16<<<dim3(MB, 2), 256, 0, stream>>>(x_bf, wt_c1, t2, NN, 128, 512, coll_b1, 1);
    k_cvt_bf16<<<cdiv((size_t)NN * 128, 256), 256, 0, stream>>>(t2, t2bf, (size_t)NN * 128);
    k_gemm_bf16<<<dim3(MB, 1), 256, 0, stream>>>(t2bf, wt_c2, collb, NN, 64, 128, coll_b2, 0);

    k_assemble<<<cdiv((size_t)NN * 577, 256), 256, 0, stream>>>(x_f32, collb, out, NN);
    k_risk_head<<<NN, 128, 0, stream>>>(t1, risk_W2, risk_b2, out);
}
